// Block_78391743086814
// MI455X (gfx1250) — matrix operand order repaired and checked by reading, not compiled, not run on hardware
//
#include <hip/hip_runtime.h>
#include <hip/hip_bf16.h>

// ---------------- problem constants ----------------
#define BB     32
#define NTOK   785
#define NPAD   800
#define CC     384
#define HH     6
#define HD     64
#define QKVC   1152           // 3*C
#define MLPH   1536           // 4*C
#define KKEEP  549            // ceil(0.7*784)
#define NNC    784            // N-1
#define NOUT   551            // 1 + KKEEP + 1
#define M1     (BB*NTOK)      // 25120 rows
#define M2     (BB*NOUT)      // 17632 rows
#define M1T32  785            // 25120/32 row-tiles
#define M2T32  551            // 17632/32 row-tiles

typedef __bf16 bf16;
typedef __attribute__((ext_vector_type(16))) __bf16 v16bf;
typedef __attribute__((ext_vector_type(8)))  float  v8f;

// gfx1250 async global->LDS path (guarded: falls back to global loads if the
// toolchain doesn't declare the builtins)
#if __has_builtin(__builtin_amdgcn_global_load_async_to_lds_b128) && \
    __has_builtin(__builtin_amdgcn_s_wait_asynccnt)
#define HAVE_ASYNC_LDS 1
#endif
#define ASG __attribute__((address_space(1)))
#define ASL __attribute__((address_space(3)))
typedef int i32x4 __attribute__((__vector_size__(16)));
typedef ASG i32x4* g_v4i_p;   // global int4*
typedef ASL i32x4* l_v4i_p;   // LDS int4*

// ---------------- WMMA helpers (CDNA5 16x16x32 bf16) ----------------
static __device__ inline v8f wmma_bf16(v16bf a, v16bf b, v8f c) {
  // 8 args: (neg_a, A, neg_b, B, c_mod, C, reuse_a, reuse_b)
  return __builtin_amdgcn_wmma_f32_16x16x32_bf16(false, a, false, b, (short)0, c, false, false);
}

// A fragment: Act is MxK row-major bf16. lane L holds row M=m0+(L&15);
// element e -> K = k0 + (L>>4)*16 + e
static __device__ inline v16bf load_a_frag(const bf16* A, int ld, int m0, int k0) {
  int lane = threadIdx.x & 31;
  int hf = lane >> 4;
  const bf16* p = A + (size_t)(m0 + (lane & 15)) * ld + k0 + hf * 16;
  v16bf a;
#pragma unroll
  for (int i = 0; i < 8; ++i) a[i] = p[i];
  p += 8;
#pragma unroll
  for (int i = 0; i < 8; ++i) a[8 + i] = p[i];
  return a;
}

// B fragment from weight W stored (out_features x in_features) row-major:
// B[k][n] = W[n][k]. lane L: n = n0+(L&15), element e -> k = k0 + (L>>4)*16 + e
static __device__ inline v16bf load_b_frag_wt(const bf16* W, int ld, int n0, int k0) {
  int lane = threadIdx.x & 31;
  const bf16* p = W + (size_t)(n0 + (lane & 15)) * ld + k0 + (lane >> 4) * 16;
  v16bf b;
#pragma unroll
  for (int i = 0; i < 16; ++i) b[i] = p[i];
  return b;
}

// B fragment from matrix stored KxN row-major (e.g. V): B[k][n] = M[k][n]
static __device__ inline v16bf load_b_frag_kn(const bf16* Bm, int ld, int k0, int n0) {
  int lane = threadIdx.x & 31;
  const bf16* p = Bm + (size_t)(k0 + (lane >> 4) * 16) * ld + n0 + (lane & 15);
  v16bf b;
#pragma unroll
  for (int i = 0; i < 16; ++i) b[i] = p[(size_t)i * ld];
  return b;
}

// ---------------- small utility kernels ----------------
__global__ void f2bf_kernel(const float* __restrict__ in, bf16* __restrict__ out, int n) {
  int i = blockIdx.x * 256 + threadIdx.x;
  if (i < n) out[i] = (bf16)in[i];
}
__global__ void zerobf_kernel(bf16* p, int n) {
  int i = blockIdx.x * 256 + threadIdx.x;
  if (i < n) p[i] = (bf16)0.0f;
}
__global__ void zerof_kernel(float* p, int n) {
  int i = blockIdx.x * 256 + threadIdx.x;
  if (i < n) p[i] = 0.0f;
}

// ---------------- LayerNorm (one wave32 per token, C=384 -> 12 elems/lane) ----
__global__ __launch_bounds__(256) void ln_kernel(const float* __restrict__ x,
                                                 const float* __restrict__ w,
                                                 const float* __restrict__ b,
                                                 bf16* __restrict__ out, int ntok) {
  int wid = threadIdx.x >> 5, lane = threadIdx.x & 31;
  int t = blockIdx.x * 8 + wid;
  if (t >= ntok) return;
  const float* row = x + (size_t)t * CC;
  float vals[12], s = 0.f;
#pragma unroll
  for (int i = 0; i < 12; ++i) { vals[i] = row[lane + i * 32]; s += vals[i]; }
#pragma unroll
  for (int sh = 16; sh > 0; sh >>= 1) s += __shfl_xor(s, sh, 32);
  float mean = s * (1.f / CC);
  float vs = 0.f;
#pragma unroll
  for (int i = 0; i < 12; ++i) { float d = vals[i] - mean; vs += d * d; }
#pragma unroll
  for (int sh = 16; sh > 0; sh >>= 1) vs += __shfl_xor(vs, sh, 32);
  float rstd = rsqrtf(vs * (1.f / CC) + 1e-5f);
#pragma unroll
  for (int i = 0; i < 12; ++i) {
    int c = lane + i * 32;
    out[(size_t)t * CC + c] = (bf16)((vals[i] - mean) * rstd * w[c] + b[c]);
  }
}

// ---------------- QKV GEMM: (25120x384)@(1152x384)^T, scatter to q/k/v -------
// one wave computes a 32x64 tile: A-frags reused x4, B-frags reused x2
__global__ __launch_bounds__(128) void gemm_qkv_kernel(const bf16* __restrict__ A,
                                                       const bf16* __restrict__ W,
                                                       const float* __restrict__ bias,
                                                       bf16* __restrict__ q,
                                                       bf16* __restrict__ k,
                                                       bf16* __restrict__ v) {
  int wid = threadIdx.x >> 5, lane = threadIdx.x & 31, hf = lane >> 4;
  int mt = blockIdx.x * 4 + wid;
  if (mt >= M1T32) return;
  int m0 = mt * 32, n0 = blockIdx.y * 64;
  v8f acc0[4] = {}, acc1[4] = {};
  for (int k0 = 0; k0 < CC; k0 += 32) {
    if (k0 + 32 < CC) __builtin_prefetch(A + (size_t)(m0 + (lane & 15)) * CC + k0 + 32, 0, 1);
    v16bf a0 = load_a_frag(A, CC, m0, k0);
    v16bf a1 = load_a_frag(A, CC, m0 + 16, k0);
#pragma unroll
    for (int t = 0; t < 4; ++t) {
      v16bf bfr = load_b_frag_wt(W, CC, n0 + t * 16, k0);
      acc0[t] = wmma_bf16(a0, bfr, acc0[t]);
      acc1[t] = wmma_bf16(a1, bfr, acc1[t]);
    }
  }
#pragma unroll
  for (int t = 0; t < 4; ++t) {
    int cc = n0 + t * 16 + (lane & 15);
    int which = cc / CC, rem = cc % CC;
    int h = rem / HD, d = rem % HD;
    bf16* dst = (which == 0) ? q : (which == 1) ? k : v;
    float bb = bias[cc];
#pragma unroll
    for (int mh = 0; mh < 2; ++mh) {
#pragma unroll
      for (int r = 0; r < 8; ++r) {
        float val = (mh ? acc1[t][r] : acc0[t][r]) + bb;
        int m = m0 + mh * 16 + r + hf * 8;
        int bt = m / NTOK, n = m % NTOK;
        dst[(((size_t)bt * HH + h) * NPAD + n) * HD + d] = (bf16)val;
      }
    }
  }
}

// ---------------- Attention: one wave per (b,h,16-query tile) ----------------
__global__ __launch_bounds__(32) void attn_kernel(const bf16* __restrict__ q,
                                                  const bf16* __restrict__ k,
                                                  const bf16* __restrict__ v,
                                                  bf16* __restrict__ attn_out,
                                                  float* __restrict__ cls_attn) {
  __shared__ float sc[16 * NPAD];   // 51200 bytes
#ifdef HAVE_ASYNC_LDS
  __shared__ bf16 vt[32 * HD];      // 4096 bytes: staged V tile
#endif
  int qt = blockIdx.x, h = blockIdx.y, b = blockIdx.z;
  int lane = threadIdx.x & 31, hf = lane >> 4;
  const size_t bh = ((size_t)b * HH + h) * NPAD * HD;
  const bf16* qb = q + bh;
  const bf16* kb = k + bh;
  const bf16* vb = v + bh;
  int m0 = qt * 16;

  v16bf aq0 = load_a_frag(qb, HD, m0, 0);
  v16bf aq1 = load_a_frag(qb, HD, m0, 32);
  const float scale = 0.125f;  // 64^-0.5

  // scores = (q*scale) @ k^T  -> LDS
  for (int jt = 0; jt < NPAD / 16; ++jt) {
    v8f acc = {};
    acc = wmma_bf16(aq0, load_b_frag_wt(kb, HD, jt * 16, 0), acc);
    acc = wmma_bf16(aq1, load_b_frag_wt(kb, HD, jt * 16, 32), acc);
    int col = jt * 16 + (lane & 15);
#pragma unroll
    for (int r = 0; r < 8; ++r) sc[(r + hf * 8) * NPAD + col] = acc[r] * scale;
  }
  __syncthreads();

  // row-wise softmax with wave32 shuffle reductions (mask padded cols)
  for (int r = 0; r < 16; ++r) {
    float mx = -1e30f;
    for (int j = lane; j < NPAD; j += 32) {
      float s = (j < NTOK) ? sc[r * NPAD + j] : -1e30f;
      mx = fmaxf(mx, s);
    }
#pragma unroll
    for (int sh = 16; sh > 0; sh >>= 1) mx = fmaxf(mx, __shfl_xor(mx, sh, 32));
    float sum = 0.f;
    for (int j = lane; j < NPAD; j += 32) {
      float p = (j < NTOK) ? __expf(sc[r * NPAD + j] - mx) : 0.f;
      sc[r * NPAD + j] = p;
      sum += p;
    }
#pragma unroll
    for (int sh = 16; sh > 0; sh >>= 1) sum += __shfl_xor(sum, sh, 32);
    float inv = 1.f / sum;
    for (int j = lane; j < NPAD; j += 32) sc[r * NPAD + j] *= inv;
  }
  __syncthreads();

  // accumulate mean-over-heads cls attention (query row 0, keys 1..784)
  if (qt == 0) {
    for (int j = 1 + lane; j < NTOK; j += 32)
      atomicAdd(&cls_attn[b * NNC + (j - 1)], sc[j] * (1.f / HH));
  }

  // out = P @ V (K padded to 800, padded probs are exactly 0)
  v8f o[4] = {};
  for (int t = 0; t < NPAD / 32; ++t) {
    int k0 = t * 32;
    v16bf a;
    int rr = lane & 15;
#pragma unroll
    for (int e = 0; e < 16; ++e) {
      int kk = k0 + hf * 16 + e;
      a[e] = (bf16)sc[rr * NPAD + kk];
    }
#ifdef HAVE_ASYNC_LDS
    // stage 32x64 bf16 V tile (4KB) via async global->LDS DMA (ASYNCcnt)
    __syncthreads();  // ensure prior ds reads of vt are done
    {
      const char* src = (const char*)(vb + (size_t)k0 * HD);
      char* dstl = (char*)vt;
#pragma unroll
      for (int c = 0; c < 8; ++c) {
        int byteoff = (lane + c * 32) * 16;
        __builtin_amdgcn_global_load_async_to_lds_b128(
            (g_v4i_p)(src + byteoff), (l_v4i_p)(dstl + byteoff), 0, 0);
      }
      __builtin_amdgcn_s_wait_asynccnt(0);
    }
#pragma unroll
    for (int x = 0; x < 4; ++x)
      o[x] = wmma_bf16(a, load_b_frag_kn(vt, HD, 0, x * 16), o[x]);
#else
#pragma unroll
    for (int x = 0; x < 4; ++x)
      o[x] = wmma_bf16(a, load_b_frag_kn(vb, HD, k0, x * 16), o[x]);
#endif
  }
#pragma unroll
  for (int x = 0; x < 4; ++x) {
    int col = h * HD + x * 16 + (lane & 15);
#pragma unroll
    for (int r = 0; r < 8; ++r) {
      int tok = m0 + r + hf * 8;
      if (tok < NTOK)
        attn_out[((size_t)b * NTOK + tok) * CC + col] = (bf16)o[x][r];
    }
  }
}

// ---------------- proj GEMM + residual: x1 = x + attn_out @ proj_w^T + b ----
__global__ __launch_bounds__(128) void gemm_proj_kernel(const bf16* __restrict__ A,
                                                        const bf16* __restrict__ W,
                                                        const float* __restrict__ bias,
                                                        const float* __restrict__ x,
                                                        float* __restrict__ x1) {
  int wid = threadIdx.x >> 5, lane = threadIdx.x & 31, hf = lane >> 4;
  int mt = blockIdx.x * 4 + wid;
  if (mt >= M1T32) return;
  int m0 = mt * 32, n0 = blockIdx.y * 64;
  v8f acc0[4] = {}, acc1[4] = {};
  for (int k0 = 0; k0 < CC; k0 += 32) {
    if (k0 + 32 < CC) __builtin_prefetch(A + (size_t)(m0 + (lane & 15)) * CC + k0 + 32, 0, 1);
    v16bf a0 = load_a_frag(A, CC, m0, k0);
    v16bf a1 = load_a_frag(A, CC, m0 + 16, k0);
#pragma unroll
    for (int t = 0; t < 4; ++t) {
      v16bf bfr = load_b_frag_wt(W, CC, n0 + t * 16, k0);
      acc0[t] = wmma_bf16(a0, bfr, acc0[t]);
      acc1[t] = wmma_bf16(a1, bfr, acc1[t]);
    }
  }
#pragma unroll
  for (int t = 0; t < 4; ++t) {
    int cc = n0 + t * 16 + (lane & 15);
    float bb = bias[cc];
#pragma unroll
    for (int mh = 0; mh < 2; ++mh) {
#pragma unroll
      for (int r = 0; r < 8; ++r) {
        size_t idx = (size_t)(m0 + mh * 16 + r + hf * 8) * CC + cc;
        x1[idx] = x[idx] + (mh ? acc1[t][r] : acc0[t][r]) + bb;
      }
    }
  }
}

// ---------------- top-K rank selection (matches jax.lax.top_k ordering) -----
__global__ __launch_bounds__(256) void topk_kernel(const float* __restrict__ cls_attn,
                                                   int* __restrict__ gidx,
                                                   int* __restrict__ keep) {
  __shared__ float vals[NNC];
  int b = blockIdx.x;
  for (int i = threadIdx.x; i < NNC; i += 256) vals[i] = cls_attn[b * NNC + i];
  __syncthreads();
  for (int i = threadIdx.x; i < NNC; i += 256) {
    float vi = vals[i];
    int rank = 0;
    for (int j = 0; j < NNC; ++j) {
      float vj = vals[j];
      rank += (vj > vi) || (vj == vi && j < i);
    }
    if (rank < KKEEP) { gidx[b * KKEEP + rank] = i; keep[b * NNC + i] = 1; }
    else keep[b * NNC + i] = 0;
  }
}

// ---------------- gather kept tokens + fused "extra" token into d_out -------
__global__ __launch_bounds__(256) void gather_kernel(const float* __restrict__ x1,
                                                     const float* __restrict__ cls_attn,
                                                     const int* __restrict__ gidx,
                                                     const int* __restrict__ keep,
                                                     float* __restrict__ xnew) {
  int n = blockIdx.x, b = blockIdx.y;
  if (n == 0) {
    for (int c = threadIdx.x; c < CC; c += 256)
      xnew[(size_t)b * NOUT * CC + c] = x1[(size_t)b * NTOK * CC + c];
  } else if (n <= KKEEP) {
    int i = gidx[b * KKEEP + (n - 1)];
    for (int c = threadIdx.x; c < CC; c += 256)
      xnew[((size_t)b * NOUT + n) * CC + c] = x1[((size_t)b * NTOK + 1 + i) * CC + c];
  } else {  // n == NOUT-1: weighted sum over non-kept tokens
    for (int c = threadIdx.x; c < CC; c += 256) {
      float s = 0.f;
      for (int j = 0; j < NNC; ++j)
        if (!keep[b * NNC + j])
          s += cls_attn[b * NNC + j] * x1[((size_t)b * NTOK + 1 + j) * CC + c];
      xnew[((size_t)b * NOUT + NOUT - 1) * CC + c] = s;
    }
  }
}

// ---------------- fc1 GEMM + exact GELU -> bf16 ------------------------------
__global__ __launch_bounds__(128) void gemm_fc1_kernel(const bf16* __restrict__ A,
                                                       const bf16* __restrict__ W,
                                                       const float* __restrict__ bias,
                                                       bf16* __restrict__ out) {
  int wid = threadIdx.x >> 5, lane = threadIdx.x & 31, hf = lane >> 4;
  int mt = blockIdx.x * 4 + wid;
  if (mt >= M2T32) return;
  int m0 = mt * 32, n0 = blockIdx.y * 64;
  v8f acc0[4] = {}, acc1[4] = {};
  for (int k0 = 0; k0 < CC; k0 += 32) {
    if (k0 + 32 < CC) __builtin_prefetch(A + (size_t)(m0 + (lane & 15)) * CC + k0 + 32, 0, 1);
    v16bf a0 = load_a_frag(A, CC, m0, k0);
    v16bf a1 = load_a_frag(A, CC, m0 + 16, k0);
#pragma unroll
    for (int t = 0; t < 4; ++t) {
      v16bf bfr = load_b_frag_wt(W, CC, n0 + t * 16, k0);
      acc0[t] = wmma_bf16(a0, bfr, acc0[t]);
      acc1[t] = wmma_bf16(a1, bfr, acc1[t]);
    }
  }
#pragma unroll
  for (int t = 0; t < 4; ++t) {
    int cc = n0 + t * 16 + (lane & 15);
    float bb = bias[cc];
#pragma unroll
    for (int mh = 0; mh < 2; ++mh) {
#pragma unroll
      for (int r = 0; r < 8; ++r) {
        float g = (mh ? acc1[t][r] : acc0[t][r]) + bb;
        g = 0.5f * g * (1.f + erff(g * 0.70710678118654752f));  // exact GELU
        out[(size_t)(m0 + mh * 16 + r + hf * 8) * MLPH + cc] = (bf16)g;
      }
    }
  }
}

// ---------------- fc2 GEMM + residual accumulate into d_out ------------------
__global__ __launch_bounds__(128) void gemm_fc2_kernel(const bf16* __restrict__ A,
                                                       const bf16* __restrict__ W,
                                                       const float* __restrict__ bias,
                                                       float* __restrict__ xnew) {
  int wid = threadIdx.x >> 5, lane = threadIdx.x & 31, hf = lane >> 4;
  int mt = blockIdx.x * 4 + wid;
  if (mt >= M2T32) return;
  int m0 = mt * 32, n0 = blockIdx.y * 64;
  v8f acc0[4] = {}, acc1[4] = {};
  for (int k0 = 0; k0 < MLPH; k0 += 32) {
    if (k0 + 32 < MLPH) __builtin_prefetch(A + (size_t)(m0 + (lane & 15)) * MLPH + k0 + 32, 0, 1);
    v16bf a0 = load_a_frag(A, MLPH, m0, k0);
    v16bf a1 = load_a_frag(A, MLPH, m0 + 16, k0);
#pragma unroll
    for (int t = 0; t < 4; ++t) {
      v16bf bfr = load_b_frag_wt(W, MLPH, n0 + t * 16, k0);
      acc0[t] = wmma_bf16(a0, bfr, acc0[t]);
      acc1[t] = wmma_bf16(a1, bfr, acc1[t]);
    }
  }
#pragma unroll
  for (int t = 0; t < 4; ++t) {
    int cc = n0 + t * 16 + (lane & 15);
    float bb = bias[cc];
#pragma unroll
    for (int mh = 0; mh < 2; ++mh) {
#pragma unroll
      for (int r = 0; r < 8; ++r) {
        size_t idx = (size_t)(m0 + mh * 16 + r + hf * 8) * CC + cc;
        xnew[idx] += (mh ? acc1[t][r] : acc0[t][r]) + bb;
      }
    }
  }
}

// ---------------- host launch ------------------------------------------------
extern "C" void kernel_launch(void* const* d_in, const int* in_sizes, int n_in,
                              void* d_out, int out_size, void* d_ws, size_t ws_size,
                              hipStream_t stream) {
  (void)in_sizes; (void)n_in; (void)out_size; (void)ws_size;
  const float* x      = (const float*)d_in[0];
  const float* n1w    = (const float*)d_in[1];
  const float* n1b    = (const float*)d_in[2];
  const float* qkvw_f = (const float*)d_in[3];
  const float* qkvb   = (const float*)d_in[4];
  const float* pw_f   = (const float*)d_in[5];
  const float* pb     = (const float*)d_in[6];
  const float* n2w    = (const float*)d_in[7];
  const float* n2b    = (const float*)d_in[8];
  const float* f1w_f  = (const float*)d_in[9];
  const float* f1b    = (const float*)d_in[10];
  const float* f2w_f  = (const float*)d_in[11];
  const float* f2b    = (const float*)d_in[12];
  float* out = (float*)d_out;

  char* ws = (char*)d_ws;
  size_t off = 0;
  auto take = [&](size_t bytes) -> void* {
    void* p = ws + off;
    off += (bytes + 255) & ~(size_t)255;
    return p;
  };
  const size_t QKVN = (size_t)BB * HH * NPAD * HD;
  bf16*  wqkv   = (bf16*)take((size_t)QKVC * CC * 2);
  bf16*  wproj  = (bf16*)take((size_t)CC * CC * 2);
  bf16*  wfc1   = (bf16*)take((size_t)MLPH * CC * 2);
  bf16*  wfc2   = (bf16*)take((size_t)CC * MLPH * 2);
  bf16*  hn     = (bf16*)take((size_t)M1 * CC * 2);
  bf16*  qb     = (bf16*)take(QKVN * 2);
  bf16*  kb     = (bf16*)take(QKVN * 2);
  bf16*  vb     = (bf16*)take(QKVN * 2);
  bf16*  attn_o = (bf16*)take((size_t)M1 * CC * 2);
  float* x1     = (float*)take((size_t)M1 * CC * 4);
  float* clsa   = (float*)take((size_t)BB * NNC * 4);
  int*   gidx   = (int*)  take((size_t)BB * KKEEP * 4);
  int*   keep   = (int*)  take((size_t)BB * NNC * 4);
  bf16*  h2n    = (bf16*)take((size_t)M2 * CC * 2);
  bf16*  f1o    = (bf16*)take((size_t)M2 * MLPH * 2);

  auto cdiv = [](int a, int b) { return (a + b - 1) / b; };

  // weights -> bf16
  f2bf_kernel<<<cdiv(QKVC * CC, 256), 256, 0, stream>>>(qkvw_f, wqkv, QKVC * CC);
  f2bf_kernel<<<cdiv(CC * CC, 256), 256, 0, stream>>>(pw_f, wproj, CC * CC);
  f2bf_kernel<<<cdiv(MLPH * CC, 256), 256, 0, stream>>>(f1w_f, wfc1, MLPH * CC);
  f2bf_kernel<<<cdiv(CC * MLPH, 256), 256, 0, stream>>>(f2w_f, wfc2, CC * MLPH);

  // zero padded qkv buffers + cls-attn accumulator (deterministic per launch)
  zerobf_kernel<<<cdiv((int)QKVN, 256), 256, 0, stream>>>(qb, (int)QKVN);
  zerobf_kernel<<<cdiv((int)QKVN, 256), 256, 0, stream>>>(kb, (int)QKVN);
  zerobf_kernel<<<cdiv((int)QKVN, 256), 256, 0, stream>>>(vb, (int)QKVN);
  zerof_kernel<<<cdiv(BB * NNC, 256), 256, 0, stream>>>(clsa, BB * NNC);

  // LN1 -> bf16
  ln_kernel<<<cdiv(M1, 8), 256, 0, stream>>>(x, n1w, n1b, hn, M1);
  // QKV GEMM
  gemm_qkv_kernel<<<dim3(cdiv(M1T32, 4), QKVC / 64), 128, 0, stream>>>(hn, wqkv, qkvb, qb, kb, vb);
  // attention
  attn_kernel<<<dim3(NPAD / 16, HH, BB), 32, 0, stream>>>(qb, kb, vb, attn_o, clsa);
  // proj + residual
  gemm_proj_kernel<<<dim3(cdiv(M1T32, 4), CC / 64), 128, 0, stream>>>(attn_o, wproj, pb, x, x1);
  // token selection
  topk_kernel<<<BB, 256, 0, stream>>>(clsa, gidx, keep);
  gather_kernel<<<dim3(NOUT, BB), 256, 0, stream>>>(x1, clsa, gidx, keep, out);
  // LN2 -> bf16 (reads pruned x from d_out)
  ln_kernel<<<cdiv(M2, 8), 256, 0, stream>>>(out, n2w, n2b, h2n, M2);
  // MLP
  gemm_fc1_kernel<<<dim3(cdiv(M2T32, 4), MLPH / 64), 128, 0, stream>>>(h2n, wfc1, f1b, f1o);
  gemm_fc2_kernel<<<dim3(cdiv(M2T32, 4), CC / 64), 128, 0, stream>>>(f1o, wfc2, f2b, out);
}
